// Loss_6433861009882
// MI455X (gfx1250) — matrix operand order repaired and checked by reading, not compiled, not run on hardware
//
#include <hip/hip_runtime.h>
#include <hip/hip_bf16.h>

typedef _Float16 v16h __attribute__((ext_vector_type(16)));
typedef _Float16 v8h  __attribute__((ext_vector_type(8)));
typedef float    v8f  __attribute__((ext_vector_type(8)));

#define NTOT   8192      // N = 2*B
#define BHALF  4096      // batch_size
#define DDIM   256
#define NCT    (NTOT / 16)   // 512 column tiles
#define SCALE  2.0f      // 1 / temperature (0.5)
#define NEG_SENT (-3.0e38f)

// ---------------------------------------------------------------------------
// Kernel 1: fp32 -> f16 conversion of [h_i ; h_j] into workspace (4 MB).
// ---------------------------------------------------------------------------
__global__ void __launch_bounds__(256) cvt_f32_to_f16(
    const float* __restrict__ hi, const float* __restrict__ hj,
    _Float16* __restrict__ out) {
  const size_t t     = (size_t)blockIdx.x * blockDim.x + threadIdx.x;
  const size_t base  = t * 8;                     // 8 | (BHALF*DDIM), no straddle
  const size_t nhalf = (size_t)BHALF * DDIM;
  const float* src = (base < nhalf) ? (hi + base) : (hj + (base - nhalf));
  float4 f0 = *(const float4*)(src);
  float4 f1 = *(const float4*)(src + 4);
  v8h o;
  o[0] = (_Float16)f0.x; o[1] = (_Float16)f0.y;
  o[2] = (_Float16)f0.z; o[3] = (_Float16)f0.w;
  o[4] = (_Float16)f1.x; o[5] = (_Float16)f1.y;
  o[6] = (_Float16)f1.z; o[7] = (_Float16)f1.w;
  *(v8h*)(out + base) = o;
}

// ---------------------------------------------------------------------------
// B tile = 32x16 f16 fragment set for the whole K=256 sweep (8 x v16h).
// Layout: lane(n = lane&15, khalf = lane>>4); per-lane K contiguous khalf*16..+15
// ---------------------------------------------------------------------------
struct BTile { v16h v[8]; };

__device__ __forceinline__ BTile load_btile(const _Float16* __restrict__ bp) {
  BTile t;
#pragma unroll
  for (int kt = 0; kt < 8; ++kt) t.v[kt] = *(const v16h*)(bp + kt * 32);
  return t;
}

__device__ __forceinline__ v8f mma_tile(const v16h a[8], const BTile& b) {
  v8f c = {0.f, 0.f, 0.f, 0.f, 0.f, 0.f, 0.f, 0.f};
#pragma unroll
  for (int kt = 0; kt < 8; ++kt) {
    c = __builtin_amdgcn_wmma_f32_16x16x32_f16(
            false, a[kt], false, b.v[kt], (short)0, c, false, false);
  }
  return c;
}

// Branchless online logsumexp update for one 16x16 sim tile (one exp/element).
__device__ __forceinline__ void softmax_update(
    const v8f& c, int col0, int r0, int half, int mlan,
    float m[8], float s[8], float& pos) {
  const int gcol = col0 + mlan;
#pragma unroll
  for (int r = 0; r < 8; ++r) {
    const int grow    = r0 + r + half * 8;      // C layout: VGPR r -> row r / r+8
    const int partner = (grow < BHALF) ? (grow + BHALF) : (grow - BHALF);
    float v  = c[r] * SCALE;
    pos     += (gcol == partner) ? v : 0.0f;
    float vm = (gcol == grow) ? NEG_SENT : v;   // drop self-sim from lse
    bool  gt   = vm > m[r];
    float newm = gt ? vm : m[r];
    float e    = __expf((gt ? m[r] : vm) - newm);  // exp(-huge) -> 0, no NaN
    s[r] = gt ? (s[r] * e + 1.0f) : (s[r] + e);
    m[r] = newm;
  }
}

// ---------------------------------------------------------------------------
// Kernel 2: fused sim-tile GEMM (WMMA f16 -> f32) + online logsumexp, with
// register double-buffered B tiles (prefetch distance = 1 tile) so the
// v_wmma stream is not serialized behind s_wait_loadcnt 0.
// ---------------------------------------------------------------------------
__global__ void __launch_bounds__(256) ntxent_fused(
    const _Float16* __restrict__ hf, float* __restrict__ partials) {
  const int lane = threadIdx.x & 31;
  const int wave = threadIdx.x >> 5;
  const int half = lane >> 4;
  const int mlan = lane & 15;
  const int rt   = blockIdx.x * 8 + wave;   // row-tile id, 0..511
  const int r0   = rt * 16;

  // ---- Load A fragments: 16 rows x 256 K (8 x v16h = 64 VGPRs) ----
  // A layout (16-bit 16x32): lane(m,half); K = half*16+0..15 (contiguous, matches B)
  const _Float16* aptr = hf + (size_t)(r0 + mlan) * DDIM + half * 16;
  v16h a[8];
#pragma unroll
  for (int kt = 0; kt < 8; ++kt) {
    v8h lo = *(const v8h*)(aptr + kt * 32);
    v8h hh = *(const v8h*)(aptr + kt * 32 + 8);
    a[kt] = __builtin_shufflevector(lo, hh, 0,1,2,3,4,5,6,7,8,9,10,11,12,13,14,15);
  }

  float m[8], s[8];
  float pos = 0.0f;
#pragma unroll
  for (int r = 0; r < 8; ++r) { m[r] = NEG_SENT; s[r] = 0.0f; }

  const _Float16* bbase = hf + (size_t)mlan * DDIM + half * 16;
  const size_t tstride = (size_t)16 * DDIM;   // f16 elems per column tile

  // Prime the ping-pong buffers.
  BTile cur = load_btile(bbase);
  BTile nxt = load_btile(bbase + tstride);

  for (int ct = 0; ct < NCT; ct += 2) {
    // ---- tile ct (cur), prefetch ct+2 into cur ----
    v8f c = mma_tile(a, cur);
    if (ct + 2 < NCT) cur = load_btile(bbase + (size_t)(ct + 2) * tstride);
    softmax_update(c, ct * 16, r0, half, mlan, m, s, pos);

    // ---- tile ct+1 (nxt), prefetch ct+3 into nxt ----
    c = mma_tile(a, nxt);
    if (ct + 3 < NCT) nxt = load_btile(bbase + (size_t)(ct + 3) * tstride);
    softmax_update(c, (ct + 1) * 16, r0, half, mlan, m, s, pos);
  }

  // ---- Merge (m,s) across the 16 lanes of each half (xor never crosses 16) --
#pragma unroll
  for (int off = 1; off < 16; off <<= 1) {
#pragma unroll
    for (int r = 0; r < 8; ++r) {
      float om  = __shfl_xor(m[r], off, 32);
      float os  = __shfl_xor(s[r], off, 32);
      float big = fmaxf(m[r], om);
      s[r] = s[r] * __expf(m[r] - big) + os * __expf(om - big);
      m[r] = big;
    }
    pos += __shfl_xor(pos, off, 32);
  }

  float acc = 0.0f;
#pragma unroll
  for (int r = 0; r < 8; ++r) acc += m[r] + logf(s[r]);
  acc -= pos;                                  // sum over 8 rows of (lse - pos)
  acc += __shfl_xor(acc, 16, 32);              // combine both halves (16 rows)
  if (lane == 0) partials[rt] = acc;
}

// ---------------------------------------------------------------------------
// Kernel 3: deterministic scalar reduction of 512 wave partials.
// ---------------------------------------------------------------------------
__global__ void reduce_partials(const float* __restrict__ partials,
                                float* __restrict__ out) {
  float acc = 0.0f;
  for (int i = threadIdx.x; i < 512; i += 32) acc += partials[i];
#pragma unroll
  for (int off = 16; off > 0; off >>= 1) acc += __shfl_xor(acc, off, 32);
  if (threadIdx.x == 0) out[0] = acc / (float)NTOT;
}

// ---------------------------------------------------------------------------
extern "C" void kernel_launch(void* const* d_in, const int* in_sizes, int n_in,
                              void* d_out, int out_size, void* d_ws, size_t ws_size,
                              hipStream_t stream) {
  const float* h_i = (const float*)d_in[0];
  const float* h_j = (const float*)d_in[1];
  // (d_in[2] = batch_size scalar; fixed at compile time)

  _Float16* hf       = (_Float16*)d_ws;                                 // 4 MB
  float*    partials = (float*)((char*)d_ws + (size_t)NTOT * DDIM * 2); // 2 KB

  cvt_f32_to_f16<<<1024, 256, 0, stream>>>(h_i, h_j, hf);
  ntxent_fused<<<64, 256, 0, stream>>>(hf, partials);
  reduce_partials<<<1, 32, 0, stream>>>(partials, (float*)d_out);
}
